// STDPConv_17583596110468
// MI455X (gfx1250) — matrix operand order repaired and checked by reading, not compiled, not run on hardware
//
#include <hip/hip_runtime.h>

typedef __attribute__((ext_vector_type(16))) _Float16 v16h;
typedef __attribute__((ext_vector_type(8)))  _Float16 v8h;
typedef __attribute__((ext_vector_type(8)))  float    v8f;

#define BATCH 16
#define CIN   64
#define COUT  128
#define HDIM  64
#define WDIM  64
#define KSZ   576   // CIN * 3 * 3
#define KTP   584   // LDS row stride in halves (16B-aligned, odd multiple of 16B stride)

// ---------------------------------------------------------------------------
// Kernel 1: fp32 -> f16 convert (x and W; OIHW flat order == [cout][576])
// ---------------------------------------------------------------------------
__global__ void to_half_kernel(const float* __restrict__ in, _Float16* __restrict__ out, int n) {
    int i = blockIdx.x * blockDim.x + threadIdx.x;
    if (i < n) out[i] = (_Float16)in[i];
}

// ---------------------------------------------------------------------------
// Kernel 2: implicit-GEMM 3x3 conv via v_wmma_f32_16x16x32_f16.
// Block = 256 threads (8 waves). Block tile: one (b, h, w0..w0+15) strip of 16
// positions, all 128 output channels (16 couts per wave). K = 576 -> 18 WMMAs,
// fully unrolled.
// ---------------------------------------------------------------------------
__global__ __launch_bounds__(256) void conv_wmma_kernel(const _Float16* __restrict__ Wh,
                                                        const _Float16* __restrict__ xh,
                                                        float* __restrict__ iout) {
    __shared__ alignas(16) _Float16 Bt[16 * KTP];   // im2col tile: [n=16 positions][K=576]

    const int tid   = threadIdx.x;
    const int wtile = blockIdx.x & 3;           // W/16 = 4 strips per row
    const int h     = (blockIdx.x >> 2) & 63;
    const int b     = blockIdx.x >> 8;
    const int w0    = wtile * 16;

    const int lane = tid & 31;
    const int wv   = tid >> 5;
    const int co0  = wv * 16;
    const int hi16 = lane >> 4;     // 0: lanes 0-15, 1: lanes 16-31
    const int l15  = lane & 15;

    // A: lane row = co0 + l15; lane K set = {kb..kb+7} U {kb+16..kb+23}, kb = kk + hi16*8
    const _Float16* arow = Wh + (co0 + l15) * KSZ + hi16 * 8;
    // warm the weight slab for this wave (gfx1250 global_prefetch_b8)
    __builtin_prefetch(arow, 0, 1);
    __builtin_prefetch(arow + 512, 0, 1);

    // ---- build im2col tile in LDS (zero-padded borders), division-free ----
    {
        const int n = tid & 15;                       // position column
        for (int ci = tid >> 4; ci < CIN; ci += 16) { // 4 input channels per thread
            const _Float16* xpl = xh + (b * CIN + ci) * (HDIM * WDIM);
            _Float16* dst = Bt + n * KTP + ci * 9;
#pragma unroll
            for (int kh = 0; kh < 3; ++kh) {
                const int hh = h + kh - 1;
                const bool hok = (hh >= 0) && (hh < HDIM);
#pragma unroll
                for (int kw = 0; kw < 3; ++kw) {
                    const int ww = w0 + n + kw - 1;
                    _Float16 v = (_Float16)0.0f;
                    if (hok && ww >= 0 && ww < WDIM) v = xpl[hh * WDIM + ww];
                    dst[kh * 3 + kw] = v;
                }
            }
        }
    }
    __syncthreads();

    // B: lane column n = l15; lane K set = {kb..kb+7} U {kb+16..kb+23}, kb = kk + hi16*8 (b128-aligned)
    const _Float16* brow = Bt + l15 * KTP + hi16 * 8;

    v8f acc = {};
#pragma unroll
    for (int kk = 0; kk < KSZ; kk += 32) {
        v8h alo = *(const v8h*)(arow + kk);        // K = kb .. kb+7
        v8h ahi = *(const v8h*)(arow + kk + 16);   // K = kb+16 .. kb+23
        v8h blo = *(const v8h*)(brow + kk);        // K = kb .. kb+7   (ds_load_b128)
        v8h bhi = *(const v8h*)(brow + kk + 16);   // K = kb+16 .. kb+23
        v16h a  = __builtin_shufflevector(alo, ahi, 0,1,2,3,4,5,6,7,8,9,10,11,12,13,14,15);
        v16h bm = __builtin_shufflevector(blo, bhi, 0,1,2,3,4,5,6,7,8,9,10,11,12,13,14,15);
        acc = __builtin_amdgcn_wmma_f32_16x16x32_f16(false, a, false, bm,
                                                     (short)0, acc, false, false);
    }

    // C/D layout: VGPR r holds (M = r + hi16*8, N = l15)
    const int w = w0 + l15;
#pragma unroll
    for (int r = 0; r < 8; ++r) {
        int co = co0 + r + hi16 * 8;
        iout[((b * COUT + co) * HDIM + h) * WDIM + w] = acc[r];
    }
}

// ---------------------------------------------------------------------------
// Kernel 3: per-output-channel max over (B,H,W) -> thre[c] = max + 1e-4
// ---------------------------------------------------------------------------
__global__ __launch_bounds__(256) void chan_max_kernel(const float* __restrict__ iout,
                                                       float* __restrict__ thre) {
    __shared__ float red[256];
    const int c = blockIdx.x;
    float m = -3.0e38f;
    for (int p = threadIdx.x; p < BATCH * HDIM * WDIM; p += 256) {
        int b  = p >> 12;           // H*W = 4096
        int hw = p & 4095;
        m = fmaxf(m, iout[(b * COUT + c) * 4096 + hw]);
    }
    red[threadIdx.x] = m;
    __syncthreads();
    for (int s = 128; s > 0; s >>= 1) {
        if (threadIdx.x < s) red[threadIdx.x] = fmaxf(red[threadIdx.x], red[threadIdx.x + s]);
        __syncthreads();
    }
    if (threadIdx.x == 0) thre[c] = red[0] + 1e-4f;
}

// ---------------------------------------------------------------------------
// Kernel 4: ASF + 10-step LIF + channel-WTA + lateral inhibition + accumulate.
// One 128-thread block (4 wave32) per spatial position; thread c = channel.
// WTA argmax(s) == lowest spiking channel: wave32 ballot + ctz, then a
// 4-entry LDS min across the block's 4 waves.
// ---------------------------------------------------------------------------
__global__ __launch_bounds__(128) void lif_wta_kernel(const float* __restrict__ iout,
                                                      const float* __restrict__ thre,
                                                      float* __restrict__ out) {
    __shared__ int wcand[4];
    const int pos = blockIdx.x;        // b*4096 + h*64 + w
    const int b   = pos >> 12;
    const int hw  = pos & 4095;
    const int c   = threadIdx.x;
    const int wv  = c >> 5;

    const float th  = thre[c];
    const float iv  = iout[(b * COUT + c) * 4096 + hw];
    const float cur = fmaxf(iv, 0.0f);
    const float asf = th / (1.0f + __expf(-(cur - 0.4f * th) * (8.0f / th)));

    const float decay = 0.99004983374917f;   // exp(-1/100)
    float mem = 0.0f, acc = 0.0f;

    for (int t = 0; t < 10; ++t) {
        mem = mem * decay + asf;
        const bool sp = (mem >= th);
        unsigned bal = __builtin_amdgcn_ballot_w32(sp);
        int cand = bal ? (wv * 32 + __builtin_ctz(bal)) : (1 << 30);
        if ((c & 31) == 0) wcand[wv] = cand;
        __syncthreads();
        const int winner = min(min(wcand[0], wcand[1]), min(wcand[2], wcand[3]));
        __syncthreads();
        const bool  swta  = sp && (c == winner);
        const float fired = (winner < (1 << 30)) ? 1.0f : 0.0f;
        mem = sp ? 0.0f : mem;                               // hard reset of all spikers
        mem -= 1.625f * th * fired * (swta ? 0.0f : 1.0f);   // inhibit non-winners at fired pos
        acc += swta ? 1.0f : 0.0f;
    }
    out[(b * COUT + c) * 4096 + hw] = acc;
}

// ---------------------------------------------------------------------------
extern "C" void kernel_launch(void* const* d_in, const int* in_sizes, int n_in,
                              void* d_out, int out_size, void* d_ws, size_t ws_size,
                              hipStream_t stream) {
    const float* x = (const float*)d_in[0];
    const float* W = (const float*)d_in[1];
    // d_in[2] = time_window (==10, baked into lif_wta_kernel)

    char* ws = (char*)d_ws;
    _Float16* Wh  = (_Float16*)(ws);                                 // 147,456 B
    _Float16* xh  = (_Float16*)(ws + 147456);                        // 8,388,608 B
    float*    io  = (float*)   (ws + 147456 + 8388608);              // 33,554,432 B
    float*    thr = (float*)   (ws + 147456 + 8388608 + 33554432);   // 512 B
    float*    out = (float*)d_out;

    const int nW = COUT * KSZ;                       // 73,728
    const int nx = BATCH * CIN * HDIM * WDIM;        // 4,194,304

    to_half_kernel<<<(nW + 255) / 256, 256, 0, stream>>>(W, Wh, nW);
    to_half_kernel<<<(nx + 255) / 256, 256, 0, stream>>>(x, xh, nx);
    conv_wmma_kernel<<<BATCH * HDIM * (WDIM / 16), 256, 0, stream>>>(Wh, xh, io);
    chan_max_kernel<<<COUT, 256, 0, stream>>>(io, thr);
    lif_wta_kernel<<<BATCH * HDIM * WDIM, 128, 0, stream>>>(io, thr, out);
}
